// Attention_74328704024929
// MI455X (gfx1250) — matrix operand order repaired and checked by reading, not compiled, not run on hardware
//
#include <hip/hip_runtime.h>
#include <hip/hip_bf16.h>

typedef __attribute__((ext_vector_type(16))) _Float16 v16h;
typedef __attribute__((ext_vector_type(8)))  _Float16 h8;
typedef __attribute__((ext_vector_type(4)))  _Float16 h4;
typedef __attribute__((ext_vector_type(8)))  float    v8f;
typedef __attribute__((ext_vector_type(4)))  float    f32x4;

#define B_  8
#define S_  2048
#define E_  768
#define N_  12
#define H_  64
#define NH_ 768          // N_*H_
#define ROWS_ 16384      // B_*S_

__device__ __forceinline__ v8f wmma_f16(v16h a, v16h b, v8f c) {
  return __builtin_amdgcn_wmma_f32_16x16x32_f16(false, a, false, b, (short)0, c,
                                                false, false);
}

// A fragment (16x32 f16) from an LDS tile laid out row-major [16][stride].
// Per ISA 7.12.2: lane L holds row L%16; K runs {kb..kb+7} and {kb+16..kb+23},
// kb = (lane&16)?8:0  -> two contiguous 8xf16 runs -> 2x ds_load_b128.
__device__ __forceinline__ v16h frag_a_lds(const _Float16* xs, int stride,
                                           int kk, int lane) {
  int row = lane & 15;
  int kb  = kk + ((lane & 16) ? 8 : 0);
  h8 a = *(const h8*)(xs + row * stride + kb);
  h8 b = *(const h8*)(xs + row * stride + kb + 16);
  v16h r;
#pragma unroll
  for (int i = 0; i < 8; ++i) { r[i] = a[i]; r[i + 8] = b[i]; }
  return r;
}

// A fragment from global f16, p -> &buf[row(lane)][kbase + ((lane&16)?8:0)]
__device__ __forceinline__ v16h frag_a_g16(const _Float16* p) {
  h8 a = *(const h8*)p;
  h8 b = *(const h8*)(p + 16);
  v16h r;
#pragma unroll
  for (int i = 0; i < 8; ++i) { r[i] = a[i]; r[i + 8] = b[i]; }
  return r;
}

// B fragment (32x16): lane holds column n = lane%16; elements run over k in
// table order.  p -> &B[k=kk+((lane&16)?8:0)][n0+lane%16], row stride given.
__device__ __forceinline__ v16h frag_b_f32row(const float* p, int stride) {
  v16h r;
#pragma unroll
  for (int i = 0; i < 16; ++i)
    r[i] = (_Float16)p[(long)(i < 8 ? i : i + 8) * stride];
  return r;
}

// B fragment, f16 gather over k in table order (V tiles: col=h=lane%16)
__device__ __forceinline__ v16h frag_b_f16row(const _Float16* p, int stride) {
  v16h r;
#pragma unroll
  for (int i = 0; i < 16; ++i) r[i] = p[(long)(i < 8 ? i : i + 8) * stride];
  return r;
}

// B fragment for K^T: lane = h (contraction), gather 16 keys at stride NH_
__device__ __forceinline__ v16h frag_bt_f16gather(const _Float16* p, int stride) {
  v16h r;
#pragma unroll
  for (int i = 0; i < 16; ++i) r[i] = p[(long)i * stride];
  return r;
}

__device__ __forceinline__ float red_max16(float v) {
  v = fmaxf(v, __shfl_xor(v, 1, 32));
  v = fmaxf(v, __shfl_xor(v, 2, 32));
  v = fmaxf(v, __shfl_xor(v, 4, 32));
  v = fmaxf(v, __shfl_xor(v, 8, 32));
  return v;
}
__device__ __forceinline__ float red_sum16(float v) {
  v += __shfl_xor(v, 1, 32);
  v += __shfl_xor(v, 2, 32);
  v += __shfl_xor(v, 4, 32);
  v += __shfl_xor(v, 8, 32);
  return v;
}

// ---------------- Kernel 1: fused QKV projection (fp32 -> f16 GEMM) --------
__global__ void __launch_bounds__(256)
qkv_kernel(const float* __restrict__ x,
           const float* __restrict__ WQ, const float* __restrict__ WK,
           const float* __restrict__ WV,
           const float* __restrict__ bQ, const float* __restrict__ bK,
           const float* __restrict__ bV,
           _Float16* __restrict__ qb, _Float16* __restrict__ kb,
           _Float16* __restrict__ vb) {
  __shared__ _Float16 xs[16 * E_];
  const int tid = threadIdx.x;
  const long mrow0 = (long)blockIdx.x * 16;

  // stage 16xE x-tile as f16 (coalesced b128 loads, b64 LDS stores)
  const float* xp = x + mrow0 * E_;
#pragma unroll
  for (int t = 0; t < 12; ++t) {
    int idx = (t * 256 + tid) * 4;
    f32x4 v = *(const f32x4*)(xp + idx);
    h4 h = { (_Float16)v[0], (_Float16)v[1], (_Float16)v[2], (_Float16)v[3] };
    *(h4*)(xs + idx) = h;
  }
  __syncthreads();

  const int lane = tid & 31, wave = tid >> 5;
  const int lrow = lane & 15;
  const int hi   = (lane & 16) ? 8 : 0;

  for (int t = wave; t < 144; t += 8) {        // 3 matrices x 48 col tiles
    const int mat = t / 48, ct = t % 48;
    const float* W   = (mat == 0) ? WQ : (mat == 1) ? WK : WV;
    const float* bia = (mat == 0) ? bQ : (mat == 1) ? bK : bV;
    _Float16*    ob  = (mat == 0) ? qb : (mat == 1) ? kb : vb;
    const int col0 = ct * 16;
    const int n = col0 >> 6, h0 = col0 & 63;
    v8f acc = {};
    for (int kk = 0; kk < E_; kk += 32) {
      v16h a = frag_a_lds(xs, E_, kk, lane);
      // W[n][e=kk+hi..][h0+lrow] : gather over e in table order
      const float* wp = W + (long)n * (E_ * H_) + (long)(kk + hi) * H_ + h0 + lrow;
      acc = wmma_f16(a, frag_b_f32row(wp, H_), acc);
    }
    const float bvv = bia[col0 + lrow];
    _Float16* orow = ob + mrow0 * NH_ + col0 + lrow;
#pragma unroll
    for (int r = 0; r < 8; ++r)
      orow[(long)(r + hi) * NH_] = (_Float16)(acc[r] + bvv);
  }
}

// ---------------- Kernel 2: flash attention, 1 wave per (b,n,16-query tile) -
__global__ void __launch_bounds__(128)
attn_kernel(const _Float16* __restrict__ qb, const _Float16* __restrict__ kbuf,
            const _Float16* __restrict__ vb, _Float16* __restrict__ zb) {
  __shared__ _Float16 Ps[4][16 * 32];          // per-wave P transpose buffer
  const int lane = threadIdx.x & 31;
  const int wave = threadIdx.x >> 5;
  const int job  = blockIdx.x * 4 + wave;      // 12288 jobs total
  const int qt   = job & 127;
  const int bn   = job >> 7;
  const int n    = bn % N_;
  const int bidx = bn / N_;
  const int q0   = qt << 4;
  const long rowbase = (long)bidx * S_;
  const int hcol = n * H_;
  const int lrow = lane & 15;
  const int hi   = (lane & 16) ? 8 : 0;

  // Q fragments, kept in VGPRs across the whole key loop
  const _Float16* qrow = qb + (rowbase + q0 + lrow) * NH_ + hcol;
  const v16h qf0 = frag_a_g16(qrow + 0  + hi);
  const v16h qf1 = frag_a_g16(qrow + 32 + hi);

  v8f o0 = {}, o1 = {}, o2 = {}, o3 = {};
  float m_[8], l_[8];
#pragma unroll
  for (int r = 0; r < 8; ++r) { m_[r] = -1e30f; l_[r] = 0.f; }

  const int cmax = (q0 + 15) >> 5;             // last (diagonal) 32-key chunk
  for (int c = 0; c <= cmax; ++c) {
    const int k0 = c << 5;
    // ---- scores S = Q K^T (16x32), 4 WMMAs ----
    v8f s0 = {}, s1 = {};
    {
      // lane = key (column); h (contraction) runs over elements, table order
      const _Float16* kp = kbuf + (rowbase + k0 + lrow) * NH_ + hcol + hi;
      s0 = wmma_f16(qf0, frag_a_g16(kp), s0);
      s1 = wmma_f16(qf0, frag_a_g16(kp + 16 * NH_), s1);
      const _Float16* kp2 = kp + 32;           // h += 32
      s0 = wmma_f16(qf1, frag_a_g16(kp2), s0);
      s1 = wmma_f16(qf1, frag_a_g16(kp2 + 16 * NH_), s1);
    }
#pragma unroll
    for (int r = 0; r < 8; ++r) { s0[r] *= 0.125f; s1[r] *= 0.125f; }
    if (c == cmax) {                           // causal mask, exact -100000
      const int kA = k0 + lrow, kB = k0 + 16 + lrow;
#pragma unroll
      for (int r = 0; r < 8; ++r) {
        const int qi = q0 + r + hi;
        if (kA > qi) s0[r] = -100000.0f;
        if (kB > qi) s1[r] = -100000.0f;
      }
    }
    // ---- online softmax (row stats live in the 16-lane groups) ----
#pragma unroll
    for (int r = 0; r < 8; ++r) {
      float mx = red_max16(fmaxf(s0[r], s1[r]));
      float mn = fmaxf(m_[r], mx);
      float al = __expf(m_[r] - mn);
      float p0 = __expf(s0[r] - mn);
      float p1 = __expf(s1[r] - mn);
      s0[r] = p0; s1[r] = p1;
      l_[r] = l_[r] * al + red_sum16(p0 + p1);
      m_[r] = mn;
      o0[r] *= al; o1[r] *= al; o2[r] *= al; o3[r] *= al;
    }
    // ---- C-layout -> A-layout transpose of P through per-wave LDS ----
    _Float16* P = Ps[wave];
#pragma unroll
    for (int r = 0; r < 8; ++r) {
      P[(r + hi) * 32 + lrow]      = (_Float16)s0[r];
      P[(r + hi) * 32 + 16 + lrow] = (_Float16)s1[r];
    }
    asm volatile("s_wait_dscnt 0" ::: "memory"); // in-wave LDS RAW fence
    const v16h pf = frag_a_lds(P, 32, 0, lane);
    // ---- PV: lane = h column, keys (contraction) gathered in table order ----
    const _Float16* vp = vb + (rowbase + k0 + hi) * NH_ + hcol + lrow;
    o0 = wmma_f16(pf, frag_b_f16row(vp, NH_),      o0);
    o1 = wmma_f16(pf, frag_b_f16row(vp + 16, NH_), o1);
    o2 = wmma_f16(pf, frag_b_f16row(vp + 32, NH_), o2);
    o3 = wmma_f16(pf, frag_b_f16row(vp + 48, NH_), o3);
  }
  // ---- finalize: z = O / l ----
  _Float16* zrow = zb + (rowbase + q0) * NH_ + hcol;
#pragma unroll
  for (int r = 0; r < 8; ++r) {
    const float inv = 1.0f / l_[r];
    const long rr = (long)(r + hi) * NH_;
    zrow[rr + 0  + lrow] = (_Float16)(o0[r] * inv);
    zrow[rr + 16 + lrow] = (_Float16)(o1[r] * inv);
    zrow[rr + 32 + lrow] = (_Float16)(o2[r] * inv);
    zrow[rr + 48 + lrow] = (_Float16)(o3[r] * inv);
  }
}

// ---------------- Kernel 3: output projection z @ W_O + b_O (fp32 out) -----
__global__ void __launch_bounds__(256)
oproj_kernel(const _Float16* __restrict__ zb, const float* __restrict__ WO,
             const float* __restrict__ bO, float* __restrict__ out) {
  __shared__ _Float16 zs[16 * NH_];
  const int tid = threadIdx.x;
  const long mrow0 = (long)blockIdx.x * 16;
  const _Float16* zp = zb + mrow0 * NH_;
#pragma unroll
  for (int t = 0; t < 6; ++t) {
    int idx = (t * 256 + tid) * 8;
    *(h8*)(zs + idx) = *(const h8*)(zp + idx);
  }
  __syncthreads();

  const int lane = tid & 31, wave = tid >> 5;
  const int lrow = lane & 15;
  const int hi   = (lane & 16) ? 8 : 0;

  for (int ct = wave; ct < 48; ct += 8) {
    const int col0 = ct * 16;
    v8f acc = {};
    for (int kk = 0; kk < NH_; kk += 32) {
      v16h a = frag_a_lds(zs, NH_, kk, lane);
      // W_O flattened is [768][768] row-major: row = n*H+h = contraction
      const float* wp = WO + (long)(kk + hi) * E_ + col0 + lrow;
      acc = wmma_f16(a, frag_b_f32row(wp, E_), acc);
    }
    const float bvv = bO[col0 + lrow];
    float* orow = out + mrow0 * E_ + col0 + lrow;
#pragma unroll
    for (int r = 0; r < 8; ++r)
      orow[(long)(r + hi) * E_] = acc[r] + bvv;
  }
}

extern "C" void kernel_launch(void* const* d_in, const int* in_sizes, int n_in,
                              void* d_out, int out_size, void* d_ws,
                              size_t ws_size, hipStream_t stream) {
  const float* x  = (const float*)d_in[0];
  const float* WQ = (const float*)d_in[1];
  const float* WK = (const float*)d_in[2];
  const float* WV = (const float*)d_in[3];
  const float* WO = (const float*)d_in[4];
  const float* bQ = (const float*)d_in[5];
  const float* bK = (const float*)d_in[6];
  const float* bV = (const float*)d_in[7];
  const float* bO = (const float*)d_in[8];
  float* out = (float*)d_out;

  const size_t NQ = (size_t)ROWS_ * NH_;       // 12.6M f16 elements each
  _Float16* qb = (_Float16*)d_ws;
  _Float16* kb = qb + NQ;
  _Float16* vb = kb + NQ;
  _Float16* zb = vb + NQ;

  qkv_kernel<<<ROWS_ / 16, 256, 0, stream>>>(x, WQ, WK, WV, bQ, bK, bV,
                                             qb, kb, vb);
  attn_kernel<<<(B_ * N_ * (S_ / 16)) / 4, 128, 0, stream>>>(qb, kb, vb, zb);
  oproj_kernel<<<ROWS_ / 16, 256, 0, stream>>>(zb, WO, bO, out);
}
